// EMD_31061203484792
// MI455X (gfx1250) — compile-verified, matrix operand order checked by reading, not run on hardware
//
#include <hip/hip_runtime.h>
#include <stdint.h>

// ---- problem constants (from reference) ----
#define N_WAY      5
#define K_SHOT     1
#define QUERY      75
#define PRJ        14
#define C_DIM      512
#define NUM_NODE   100
#define NODE_PAD   112                      // 7 * 16, zero-padded rows
#define N_SUPPORT  (N_WAY * K_SHOT * PRJ)   // 70
#define NQ         (N_WAY * QUERY)          // 375
#define NPAIR      (NQ * N_WAY)             // 1875
#define SINK_ITERS 50
#define SAMPLE_ELEMS (C_DIM * NUM_NODE)     // 51200

#define KC         64                        // K-chunk (channels) staged in LDS
#define LROW       72                        // LDS row stride in bf16 (144 B, 16B aligned)

typedef __attribute__((ext_vector_type(16))) __bf16 v16bf;
typedef __attribute__((ext_vector_type(8)))  float  v8f;
typedef __attribute__((ext_vector_type(4)))  int    v4i;
typedef __attribute__((address_space(1))) v4i gv4i;  // global-AS int4
typedef __attribute__((address_space(3))) v4i lv4i;  // LDS-AS int4

#if defined(__gfx1250__) && __has_builtin(__builtin_amdgcn_global_load_async_to_lds_b128) && __has_builtin(__builtin_amdgcn_s_wait_asynccnt)
#define USE_ASYNC_LDS 1
#else
#define USE_ASYNC_LDS 0
#endif

static __device__ __forceinline__ unsigned short f32_to_bf16(float x) {
    union { float f; uint32_t u; } v; v.f = x;
    uint32_t r = v.u + 0x7FFFu + ((v.u >> 16) & 1u);   // round-to-nearest-even
    return (unsigned short)(r >> 16);
}

// ---- K1: proto[n][c][node] = mean over 14 shots ----
__global__ void k_proto(const float* __restrict__ feat, float* __restrict__ proto) {
    int idx = blockIdx.x * blockDim.x + threadIdx.x;
    if (idx >= N_WAY * SAMPLE_ELEMS) return;
    int n = idx / SAMPLE_ELEMS;
    int rem = idx % SAMPLE_ELEMS;
    float s = 0.f;
    for (int k = 0; k < PRJ; ++k)
        s += feat[(size_t)(n * PRJ + k) * SAMPLE_ELEMS + rem];
    proto[idx] = s * (1.0f / PRJ);
}

// ---- K2: proto_avg[n][c], query_avg[m][c] = spatial means ----
__global__ void k_avg(const float* __restrict__ feat, const float* __restrict__ proto,
                      float* __restrict__ pavg, float* __restrict__ qavg) {
    int idx = blockIdx.x * blockDim.x + threadIdx.x;
    if (idx >= (N_WAY + NQ) * C_DIM) return;
    const float* src; float* dst;
    if (idx < N_WAY * C_DIM) {
        src = proto + (size_t)idx * NUM_NODE;
        dst = pavg + idx;
    } else {
        int q = idx - N_WAY * C_DIM;
        int m = q / C_DIM, c = q % C_DIM;
        src = feat + ((size_t)(N_SUPPORT + m) * C_DIM + c) * NUM_NODE;
        dst = qavg + q;
    }
    float s = 0.f;
    for (int p = 0; p < NUM_NODE; ++p) s += src[p];
    *dst = s * (1.0f / NUM_NODE);
}

// ---- K3: channel-center per (sample,node); emit bf16 [sample][node_pad][c] + sq-norm ----
// One block of 128 threads per (sample, padded-node); thread t owns channels [4t,4t+4).
// Padded nodes (>=100) are written as zero rows so the GEMM needs no validity checks.
__global__ void k_center(const float* __restrict__ feat, const float* __restrict__ proto,
                         unsigned short* __restrict__ qb, unsigned short* __restrict__ pb,
                         float* __restrict__ qn, float* __restrict__ pn) {
    int sidx = blockIdx.x / NODE_PAD;   // 0..379 (375 queries, then 5 protos)
    int node = blockIdx.x % NODE_PAD;
    const float* src; unsigned short* dstb; float* dstn;
    if (sidx < NQ) {
        src  = feat + (size_t)(N_SUPPORT + sidx) * SAMPLE_ELEMS;
        dstb = qb + ((size_t)sidx * NODE_PAD + node) * C_DIM;
        dstn = qn + sidx * NUM_NODE + node;
    } else {
        int n = sidx - NQ;
        src  = proto + (size_t)n * SAMPLE_ELEMS;
        dstb = pb + ((size_t)n * NODE_PAD + node) * C_DIM;
        dstn = pn + n * NUM_NODE + node;
    }
    int t = threadIdx.x;
    if (node >= NUM_NODE) {             // block-uniform: zero pad row, no barriers needed
        uint2 z = {0u, 0u};
        *(uint2*)(dstb + 4 * t) = z;
        return;
    }
    float v[4]; float s = 0.f;
    for (int k = 0; k < 4; ++k) {
        v[k] = src[(size_t)(4 * t + k) * NUM_NODE + node];
        s += v[k];
    }
    __shared__ float red[128];
    red[t] = s; __syncthreads();
    for (int off = 64; off > 0; off >>= 1) { if (t < off) red[t] += red[t + off]; __syncthreads(); }
    float mu = red[0] * (1.0f / C_DIM);
    __syncthreads();
    float ss = 0.f; unsigned short o[4];
    for (int k = 0; k < 4; ++k) { float c = v[k] - mu; ss += c * c; o[k] = f32_to_bf16(c); }
    uint2 pk;
    pk.x = (uint32_t)o[0] | ((uint32_t)o[1] << 16);
    pk.y = (uint32_t)o[2] | ((uint32_t)o[3] << 16);
    *(uint2*)(dstb + 4 * t) = pk;                  // coalesced 8B per thread
    red[t] = ss; __syncthreads();
    for (int off = 64; off > 0; off >>= 1) { if (t < off) red[t] += red[t + off]; __syncthreads(); }
    if (t == 0) *dstn = red[0];
}

// ---- K4: log of normalized row marginal ww1 (uses raw query, proto_avg) ----
__global__ void k_logr(const float* __restrict__ feat, const float* __restrict__ pavg,
                       float* __restrict__ logr) {
    int pair = blockIdx.x; int i = pair / N_WAY, j = pair % N_WAY;
    int t = threadIdx.x;                          // 128 threads, 100 active
    __shared__ float red[128];
    float w = 0.f;
    if (t < NUM_NODE) {
        const float* q  = feat + (size_t)(N_SUPPORT + i) * SAMPLE_ELEMS;
        const float* pa = pavg + j * C_DIM;
        float dot = 0.f;
        for (int c = 0; c < C_DIM; ++c) dot += q[(size_t)c * NUM_NODE + t] * pa[c];
        w = fmaxf(dot, 0.f) + 0.001f + 1e-5f;     // relu+0.001, then +1e-5
    }
    red[t] = w; __syncthreads();
    for (int off = 64; off > 0; off >>= 1) { if (t < off) red[t] += red[t + off]; __syncthreads(); }
    float norm = __logf((float)NUM_NODE / red[0]);
    if (t < NUM_NODE) logr[(size_t)pair * NUM_NODE + t] = __logf(w) + norm;
}

// ---- K5: log of normalized column marginal ww2 (uses proto, query_avg) ----
__global__ void k_logc(const float* __restrict__ proto, const float* __restrict__ qavg,
                       float* __restrict__ logc) {
    int pair = blockIdx.x; int i = pair / N_WAY, j = pair % N_WAY;
    int t = threadIdx.x;
    __shared__ float red[128];
    float w = 0.f;
    if (t < NUM_NODE) {
        const float* pr = proto + (size_t)j * SAMPLE_ELEMS;
        const float* qa = qavg + i * C_DIM;
        float dot = 0.f;
        for (int c = 0; c < C_DIM; ++c) dot += pr[(size_t)c * NUM_NODE + t] * qa[c];
        w = fmaxf(dot, 0.f) + 0.001f + 1e-5f;
    }
    red[t] = w; __syncthreads();
    for (int off = 64; off > 0; off >>= 1) { if (t < off) red[t] += red[t + off]; __syncthreads(); }
    float norm = __logf((float)NUM_NODE / red[0]);
    if (t < NUM_NODE) logc[(size_t)pair * NUM_NODE + t] = __logf(w) + norm;
}

// ---- K6: fused WMMA cost + log-domain Sinkhorn + logits; 1 block per (query,class) ----
// Dynamic LDS layout (bytes):
//   Cn   [100*101] f32  : 40400 (padded to 40448)
//   qs   [112*72]  bf16 : 16128
//   ps   [112*72]  bf16 : 16128
//   fA,gA,lr,lc [100]f32: 1600 (padded 1664)
//   red  [256] f32      : 1024
#define SM_CN   0
#define SM_QS   40448
#define SM_PS   (SM_QS + 16128)
#define SM_F    (SM_PS + 16128)
#define SM_G    (SM_F + 400)
#define SM_LR   (SM_G + 400)
#define SM_LC   (SM_LR + 400)
#define SM_RED  (SM_LC + 464)               // re-align to 16
#define SM_TOTAL (SM_RED + 1024)

__global__ void __launch_bounds__(256) k_emd(
    const unsigned short* __restrict__ qb, const unsigned short* __restrict__ pb,
    const float* __restrict__ qn, const float* __restrict__ pn,
    const float* __restrict__ logr, const float* __restrict__ logc,
    float* __restrict__ logits)
{
    constexpr int STRIDE = 101;                   // odd stride: conflict-free column access
    extern __shared__ char smem[];
    float* Cn = (float*)(smem + SM_CN);
    unsigned short* qs = (unsigned short*)(smem + SM_QS);
    unsigned short* ps = (unsigned short*)(smem + SM_PS);
    float* fA = (float*)(smem + SM_F);
    float* gA = (float*)(smem + SM_G);
    float* lr = (float*)(smem + SM_LR);
    float* lc = (float*)(smem + SM_LC);
    float* red = (float*)(smem + SM_RED);

    int pair = blockIdx.x;
    int i = pair / N_WAY, j = pair % N_WAY;
    int t = threadIdx.x;
    int wave = t >> 5, lane = t & 31;             // wave32

    const unsigned short* qrow = qb + (size_t)i * NODE_PAD * C_DIM;
    const unsigned short* prow = pb + (size_t)j * NODE_PAD * C_DIM;

    // ---- Phase 1: cross via bf16 WMMA over 7x7 grid of 16x16 tiles, K chunked in LDS ----
    int mnode = lane & 15;
    int base0 = (lane < 16) ? 0 : 8;              // 16-bit A/B lane swizzle (ISA 7.12.2)
    v8f accT[7];
    #pragma unroll
    for (int s = 0; s < 7; ++s) accT[s] = v8f{};

    constexpr int SEGS = NODE_PAD * (KC / 8);     // 112 rows * 8 x b128 segments = 896
    for (int kc = 0; kc < C_DIM; kc += KC) {
        // cooperative chunk copy: q and p, 2*896 b128 transfers over 256 threads
        for (int s = t; s < 2 * SEGS; s += 256) {
            int which = (s >= SEGS) ? 1 : 0;
            int e = which ? (s - SEGS) : s;
            int row = e >> 3, seg = e & 7;
            const unsigned short* g = (which ? prow : qrow) + (size_t)row * C_DIM + kc + seg * 8;
            unsigned short* l = (which ? ps : qs) + row * LROW + seg * 8;
#if USE_ASYNC_LDS
            __builtin_amdgcn_global_load_async_to_lds_b128(
                (gv4i*)(void*)g, (lv4i*)(void*)l, 0, 0);
#else
            *(uint4*)l = *(const uint4*)g;
#endif
        }
#if USE_ASYNC_LDS
        __builtin_amdgcn_s_wait_asynccnt(0);
#endif
        __syncthreads();

        #pragma unroll
        for (int slot = 0; slot < 7; ++slot) {
            int tile = wave + slot * 8;
            if (tile < 49) {
                int tm = tile / 7, tn = tile % 7;
                const unsigned short* aL = qs + (tm * 16 + mnode) * LROW + base0;
                const unsigned short* bL = ps + (tn * 16 + mnode) * LROW + base0;
                #pragma unroll
                for (int ks = 0; ks < KC; ks += 32) {
                    union { v16bf v; uint4 u[2]; } A, B;
                    A.u[0] = *(const uint4*)(aL + ks);
                    A.u[1] = *(const uint4*)(aL + ks + 16);
                    B.u[0] = *(const uint4*)(bL + ks);
                    B.u[1] = *(const uint4*)(bL + ks + 16);
                    accT[slot] = __builtin_amdgcn_wmma_f32_16x16x32_bf16(
                        false, A.v, false, B.v, (short)0, accT[slot], false, false);
                }
            }
        }
        __syncthreads();                          // chunk buffers reused next iteration
    }

    // store: cost[a][b] = qn[a] + pn[b] - 2*cross, C/D layout row = r+8*(lane>=16), col = lane%16
    #pragma unroll
    for (int slot = 0; slot < 7; ++slot) {
        int tile = wave + slot * 8;
        if (tile < 49) {
            int tm = tile / 7, tn = tile % 7;
            #pragma unroll
            for (int r = 0; r < 8; ++r) {
                int a = tm * 16 + r + ((lane >= 16) ? 8 : 0);
                int b = tn * 16 + (lane & 15);
                if (a < NUM_NODE && b < NUM_NODE)
                    Cn[a * STRIDE + b] = qn[i * NUM_NODE + a] + pn[j * NUM_NODE + b]
                                         - 2.0f * accT[slot][r];
            }
        }
    }
    __syncthreads();

    // ---- Phase 2: max-normalize cost (matches reference scale trick) ----
    float mx = -1e30f;
    for (int e = t; e < NUM_NODE * NUM_NODE; e += 256) {
        int a = e / NUM_NODE, b = e % NUM_NODE;
        mx = fmaxf(mx, Cn[a * STRIDE + b]);
    }
    red[t] = mx; __syncthreads();
    for (int off = 128; off > 0; off >>= 1) { if (t < off) red[t] = fmaxf(red[t], red[t + off]); __syncthreads(); }
    float scale = red[0] + 1e-6f;
    float inv = 1.0f / scale;
    for (int e = t; e < NUM_NODE * NUM_NODE; e += 256) {
        int a = e / NUM_NODE, b = e % NUM_NODE;
        Cn[a * STRIDE + b] *= inv;
    }
    if (t < NUM_NODE) {
        fA[t] = 0.f; gA[t] = 0.f;
        lr[t] = logr[(size_t)pair * NUM_NODE + t];
        lc[t] = logc[(size_t)pair * NUM_NODE + t];
    }
    __syncthreads();

    // ---- Phase 3: log-domain Sinkhorn, 50 iterations ----
    const float eps = 0.05f, ieps = 20.0f;
    for (int it = 0; it < SINK_ITERS; ++it) {
        if (t < NUM_NODE) {                       // f-update: row t, sweep columns
            const float* row = &Cn[t * STRIDE];
            float m = -1e30f;
            for (int b = 0; b < NUM_NODE; ++b) m = fmaxf(m, gA[b] - row[b]);
            float s = 0.f;
            for (int b = 0; b < NUM_NODE; ++b) s += __expf((gA[b] - row[b] - m) * ieps);
            fA[t] = eps * lr[t] - m - eps * __logf(s);
        }
        __syncthreads();
        if (t < NUM_NODE) {                       // g-update: column t, sweep rows
            float m = -1e30f;
            for (int a = 0; a < NUM_NODE; ++a) m = fmaxf(m, fA[a] - Cn[a * STRIDE + t]);
            float s = 0.f;
            for (int a = 0; a < NUM_NODE; ++a) s += __expf((fA[a] - Cn[a * STRIDE + t] - m) * ieps);
            gA[t] = eps * lc[t] - m - eps * __logf(s);
        }
        __syncthreads();
    }

    // ---- Phase 4: logits = sum(sim * flow) * 12.5/NUM_NODE ----
    float acc = 0.f;
    for (int e = t; e < NUM_NODE * NUM_NODE; e += 256) {
        int a = e / NUM_NODE, b = e % NUM_NODE;
        float c = Cn[a * STRIDE + b];
        float flow = __expf((fA[a] + gA[b] - c) * ieps);
        float sim = 1.0f - c * scale;             // sim on the UNnormalized cost
        acc += sim * flow;
    }
    red[t] = acc; __syncthreads();
    for (int off = 128; off > 0; off >>= 1) { if (t < off) red[t] += red[t + off]; __syncthreads(); }
    if (t == 0) logits[pair] = red[0] * (12.5f / NUM_NODE);
}

// ---- K7: log-softmax over classes + NLL loss ----
__global__ void k_softmax(const float* __restrict__ logits, const int* __restrict__ label,
                          float* __restrict__ out) {
    int t = threadIdx.x;                          // 512 threads, 375 active
    __shared__ float red[512];
    float lossp = 0.f;
    if (t < NQ) {
        float l[N_WAY]; float mx = -1e30f;
        for (int j = 0; j < N_WAY; ++j) { l[j] = logits[t * N_WAY + j]; mx = fmaxf(mx, l[j]); }
        float s = 0.f;
        for (int j = 0; j < N_WAY; ++j) s += __expf(l[j] - mx);
        float lse = mx + __logf(s);
        for (int j = 0; j < N_WAY; ++j) out[t * N_WAY + j] = l[j] - lse;
        lossp = -(l[label[t]] - lse);
    }
    red[t] = lossp; __syncthreads();
    for (int off = 256; off > 0; off >>= 1) { if (t < off) red[t] += red[t + off]; __syncthreads(); }
    if (t == 0) out[NQ * N_WAY] = red[0] / (float)NQ;
}

extern "C" void kernel_launch(void* const* d_in, const int* in_sizes, int n_in,
                              void* d_out, int out_size, void* d_ws, size_t ws_size,
                              hipStream_t stream) {
    (void)in_sizes; (void)n_in; (void)out_size; (void)ws_size;
    const float* feat = (const float*)d_in[0];
    const int* label  = (const int*)d_in[1];
    float* out = (float*)d_out;

    // workspace carve-out (~47 MB total)
    char* ws = (char*)d_ws;
    size_t off = 0;
    auto alloc = [&](size_t bytes) { void* p = ws + off; off = (off + bytes + 255) & ~(size_t)255; return p; };
    float* proto = (float*)alloc(sizeof(float) * (size_t)N_WAY * SAMPLE_ELEMS);
    float* pavg  = (float*)alloc(sizeof(float) * (size_t)N_WAY * C_DIM);
    float* qavg  = (float*)alloc(sizeof(float) * (size_t)NQ * C_DIM);
    unsigned short* qb = (unsigned short*)alloc(sizeof(unsigned short) * (size_t)NQ * NODE_PAD * C_DIM);
    unsigned short* pb = (unsigned short*)alloc(sizeof(unsigned short) * (size_t)N_WAY * NODE_PAD * C_DIM);
    float* qn   = (float*)alloc(sizeof(float) * (size_t)NQ * NUM_NODE);
    float* pn   = (float*)alloc(sizeof(float) * (size_t)N_WAY * NUM_NODE);
    float* logr = (float*)alloc(sizeof(float) * (size_t)NPAIR * NUM_NODE);
    float* logc = (float*)alloc(sizeof(float) * (size_t)NPAIR * NUM_NODE);
    float* lg   = (float*)alloc(sizeof(float) * (size_t)NPAIR);

    k_proto <<<(N_WAY * SAMPLE_ELEMS + 255) / 256, 256, 0, stream>>>(feat, proto);
    k_avg   <<<((N_WAY + NQ) * C_DIM + 255) / 256, 256, 0, stream>>>(feat, proto, pavg, qavg);
    k_center<<<(NQ + N_WAY) * NODE_PAD, 128, 0, stream>>>(feat, proto, qb, pb, qn, pn);
    k_logr  <<<NPAIR, 128, 0, stream>>>(feat, pavg, logr);
    k_logc  <<<NPAIR, 128, 0, stream>>>(proto, qavg, logc);
    k_emd   <<<NPAIR, 256, SM_TOTAL, stream>>>(qb, pb, qn, pn, logr, logc, lg);
    k_softmax<<<1, 512, 0, stream>>>(lg, label, out);
}
